// GatedEncoderLayer_63041529971397
// MI455X (gfx1250) — compile-verified, matrix operand order checked by reading, not run on hardware
//
#include <hip/hip_runtime.h>
#include <hip/hip_bf16.h>

// ---------------------------------------------------------------------------
// GatedEncoderLayer for MI455X (gfx1250, wave32, WMMA bf16 + pipelined TDM)
// ---------------------------------------------------------------------------
#define B_    4
#define S_    2048
#define D_    512
#define H_    8
#define HD_   64
#define FC_   2048
#define STR_  512   // STRIDE = S * 0.25

typedef __attribute__((ext_vector_type(16))) __bf16 bf16x16;
typedef __attribute__((ext_vector_type(8)))  __bf16 bf16x8;
typedef __attribute__((ext_vector_type(8)))  float  f32x8;
typedef __attribute__((ext_vector_type(4)))  unsigned int u32x4;
typedef __attribute__((ext_vector_type(4)))  int    i32x4;
typedef __attribute__((ext_vector_type(8)))  int    i32x8;

#ifndef __has_builtin
#define __has_builtin(x) 0
#endif
#if __has_builtin(__builtin_amdgcn_tensor_load_to_lds) && \
    __has_builtin(__builtin_amdgcn_s_wait_tensorcnt)
#define HAVE_TDM 1
#else
#define HAVE_TDM 0
#endif

__device__ __forceinline__ f32x8 wmma_bf16(bf16x16 a, bf16x16 b, f32x8 c) {
    return __builtin_amdgcn_wmma_f32_16x16x32_bf16(false, a, false, b,
                                                   (short)0, c, false, false);
}

// Load one 16-lane-row fragment (A- or B-style) for 16x16x32 bf16 WMMA.
// lane<16 holds K [k,k+8),[k+16,k+24); lane>=16 holds [k+8,k+16),[k+24,k+32)
__device__ __forceinline__ bf16x16 load_frag(const __bf16* rowbase, int k, int hf) {
    bf16x8 lo = *(const bf16x8*)(rowbase + k + 8 * hf);
    bf16x8 hi = *(const bf16x8*)(rowbase + k + 16 + 8 * hf);
    return __builtin_shufflevector(lo, hi, 0,1,2,3,4,5,6,7,8,9,10,11,12,13,14,15);
}

// ---------------------------------------------------------------------------
__global__ void cvt_bf16(const float* __restrict__ in, __bf16* __restrict__ out, int n) {
    int i = blockIdx.x * 256 + threadIdx.x;
    if (i < n) out[i] = (__bf16)in[i];
}

__global__ void zero_f32(float* p, int n) {
    int i = blockIdx.x * 256 + threadIdx.x;
    if (i < n) p[i] = 0.0f;
}

// ---------------------------------------------------------------------------
// GEMM: C[M,N] = A[M,K](bf16) @ W[N,K]^T(bf16) + bias, optional SiLU,
// optional bf16 mirror.
// Block = 4 waves; wave tile 64(M)x64(N); block tile 256(M)x64(N).
// All 4 waves share the same 64 W-rows: W 64x32 sub-tiles are staged in a
// double-buffered LDS region by the Tensor Data Mover, pipelined one chunk
// ahead of the WMMA stream.  __launch_bounds__(128,1) -> full VGPR budget,
// no accumulator spills.
__global__ __launch_bounds__(128, 1)
void gemm_bf16(const __bf16* __restrict__ A, const __bf16* __restrict__ W,
               const float* __restrict__ bias, float* __restrict__ C,
               __bf16* __restrict__ Cb, int M, int N, int K, int act) {
    __shared__ __bf16 wtile[2][64 * 32];       // 2 x 4 KB staging buffers
    int lane = threadIdx.x & 31;
    int wave = threadIdx.x >> 5;
    int hf = lane >> 4, ln = lane & 15;
    int m0 = blockIdx.y * 256 + wave * 64;
    int n0 = blockIdx.x * 64;
    const __bf16* arow[4];
#pragma unroll
    for (int mt = 0; mt < 4; ++mt)
        arow[mt] = A + (size_t)(m0 + mt * 16 + ln) * K;
    f32x8 acc[4][4] = {};

#if HAVE_TDM
    auto tdm_load = [&](int k, int buf) {
        // ---- Tensor DMA descriptor (D#): 64 rows x 32 cols bf16 ----
        unsigned long long ga = (unsigned long long)(uintptr_t)(W + (size_t)n0 * K + k);
        unsigned int la = (unsigned int)(uintptr_t)(void*)&wtile[buf][0];
        u32x4 g0;
        g0[0] = 1u;                               // count = 1 (valid D#)
        g0[1] = la;                               // lds_addr
        g0[2] = (unsigned int)ga;                 // global_addr[31:0]
        g0[3] = (unsigned int)((ga >> 32) & 0x01FFFFFFu) | (2u << 30); // addr[56:32] | type=2
        i32x8 g1;
        g1[0] = 0x00010000;                       // wg_mask=0, data_size=1 (2 bytes)
        g1[1] = (int)((K & 0xFFFF) << 16);        // tensor_dim0[15:0]
        g1[2] = (int)(((unsigned)K >> 16) | (((unsigned)N & 0xFFFF) << 16));
        g1[3] = (int)(((unsigned)N >> 16) | (32u << 16));  // tile_dim0 = 32
        g1[4] = 64;                               // tile_dim1 = 64 (tile_dim2 = 0)
        g1[5] = K;                                // tensor_dim0_stride[31:0]
        g1[6] = 0;
        g1[7] = 0;
        i32x4 z4 = {0, 0, 0, 0};
#if defined(__clang_major__) && (__clang_major__ >= 23)
        i32x8 z8 = {0, 0, 0, 0, 0, 0, 0, 0};
        __builtin_amdgcn_tensor_load_to_lds(g0, g1, z4, z4, z8, 0);
#else
        __builtin_amdgcn_tensor_load_to_lds(g0, g1, z4, z4, 0);
#endif
    };
    if (wave == 0) {
        tdm_load(0, 0);
        __builtin_amdgcn_s_wait_tensorcnt((short)0);
    }
    __syncthreads();
#else
    auto coop_stage = [&](int k, int buf) {
        for (int idx = threadIdx.x; idx < 256; idx += 128) {
            int r = idx >> 2, c = (idx & 3) * 8;
            *(bf16x8*)&wtile[buf][r * 32 + c] =
                *(const bf16x8*)(W + (size_t)(n0 + r) * K + k + c);
        }
    };
    coop_stage(0, 0);
    __syncthreads();
#endif

    int nch = K / 32;
    for (int i = 0; i < nch; ++i) {
        int k = i * 32;
        int cur = i & 1, nxt = cur ^ 1;
#if HAVE_TDM
        if (i + 1 < nch && wave == 0) tdm_load(k + 32, nxt);   // pipeline next tile
#else
        if (i + 1 < nch) coop_stage(k + 32, nxt);
#endif
        bf16x16 af[4];
#pragma unroll
        for (int mt = 0; mt < 4; ++mt) {
            if (k + 64 < K) __builtin_prefetch(arow[mt] + k + 64, 0, 1); // global_prefetch_b8
            af[mt] = load_frag(arow[mt], k, hf);
        }
#pragma unroll
        for (int t = 0; t < 4; ++t) {
            const __bf16* wr = &wtile[cur][(t * 16 + ln) * 32];
            bf16x8 lo = *(const bf16x8*)(wr + 8 * hf);
            bf16x8 hi = *(const bf16x8*)(wr + 16 + 8 * hf);
            bf16x16 wf = __builtin_shufflevector(lo, hi,
                             0,1,2,3,4,5,6,7,8,9,10,11,12,13,14,15);
#pragma unroll
            for (int mt = 0; mt < 4; ++mt)
                acc[mt][t] = wmma_bf16(af[mt], wf, acc[mt][t]);
        }
#if HAVE_TDM
        if (wave == 0) __builtin_amdgcn_s_wait_tensorcnt((short)0);  // next tile landed
#endif
        __syncthreads();
    }
#pragma unroll
    for (int mt = 0; mt < 4; ++mt)
#pragma unroll
        for (int t = 0; t < 4; ++t)
#pragma unroll
            for (int i = 0; i < 8; ++i) {
                int row = m0 + mt * 16 + i + 8 * hf;
                int col = n0 + t * 16 + ln;
                float v = acc[mt][t][i] + bias[col];
                if (act == 1) v = v / (1.0f + __expf(-v));   // SiLU
                C[(size_t)row * N + col] = v;
                if (Cb) Cb[(size_t)row * N + col] = (__bf16)v;
            }
}

// ---------------------------------------------------------------------------
// qkv f32 [B,S,3*D] -> qb,kb bf16 [B,H,S,64] ; vt bf16 [B,H,64,S]
__global__ void pack_qkv(const float* __restrict__ qkv, __bf16* __restrict__ qb,
                         __bf16* __restrict__ kb, __bf16* __restrict__ vt) {
    int i = blockIdx.x * 256 + threadIdx.x;
    if (i >= B_ * S_ * H_ * HD_) return;
    int hd = i & 63;
    int t  = i >> 6;
    int h  = t & 7;  t >>= 3;
    int s  = t & (S_ - 1);
    int b  = t / S_;
    size_t src = ((size_t)b * S_ + s) * (3 * D_) + h * HD_ + hd;
    int bh = b * H_ + h;
    qb[((size_t)bh * S_ + s) * HD_ + hd] = (__bf16)qkv[src];
    kb[((size_t)bh * S_ + s) * HD_ + hd] = (__bf16)qkv[src + D_];
    vt[((size_t)bh * HD_ + hd) * S_ + s] = (__bf16)qkv[src + 2 * D_];
}

__global__ void compute_lens(const unsigned char* __restrict__ pad, int* __restrict__ lens) {
    __shared__ int r[256];
    int b = blockIdx.x, t = threadIdx.x, sum = 0;
    for (int i = t; i < S_; i += 256) sum += pad[b * S_ + i] ? 1 : 0;
    r[t] = sum; __syncthreads();
    for (int s = 128; s > 0; s >>= 1) { if (t < s) r[t] += r[t + s]; __syncthreads(); }
    if (t == 0) lens[b] = S_ - r[0];
}

// ---------------------------------------------------------------------------
// Attention pass 1: per (b,h, 16-query tile) online softmax stats.
// logits^T = Ktile(16 x hd) x Q^T(hd x 16q) -> rows = keys, cols = queries.
__global__ __launch_bounds__(32, 1)
void attn_stats(const __bf16* __restrict__ qb, const __bf16* __restrict__ kb,
                const int* __restrict__ lens,
                float* __restrict__ mrow, float* __restrict__ srow) {
    int wid = blockIdx.x;                     // B*H*(S/16)
    int qt  = wid & (S_ / 16 - 1);
    int bh  = wid / (S_ / 16);
    int b   = bh / H_;
    int lane = threadIdx.x, hf = lane >> 4, ln = lane & 15;
    int q0  = qt * 16;
    int len = lens[b];
    const __bf16* qrow = qb + ((size_t)bh * S_ + q0 + ln) * HD_;
    bf16x16 bq0 = load_frag(qrow, 0,  hf);
    bf16x16 bq1 = load_frag(qrow, 32, hf);
    const __bf16* kbase = kb + (size_t)bh * S_ * HD_;
    float m = -1e30f, s = 0.0f;
    for (int kc = 0; kc < S_; kc += 16) {
        const __bf16* krow = kbase + (size_t)(kc + ln) * HD_;
        f32x8 acc = {};
        acc = wmma_bf16(load_frag(krow, 0,  hf), bq0, acc);
        acc = wmma_bf16(load_frag(krow, 32, hf), bq1, acc);
        float vals[8]; float tmax = -1e30f;
#pragma unroll
        for (int i = 0; i < 8; ++i) {
            int key = kc + i + 8 * hf;
            float v = acc[i] * 0.125f;        // 1/sqrt(64)
            if (key >= len) v = -1e30f;
            vals[i] = v; tmax = fmaxf(tmax, v);
        }
        float nm = fmaxf(m, tmax);
        float add = 0.0f;
#pragma unroll
        for (int i = 0; i < 8; ++i) add += __expf(vals[i] - nm);
        s = s * __expf(m - nm) + add;
        m = nm;
    }
    float mo = __shfl_xor(m, 16, 32), so = __shfl_xor(s, 16, 32);
    float nm = fmaxf(m, mo);
    float ns = s * __expf(m - nm) + so * __expf(mo - nm);
    if (lane < 16) {
        mrow[(size_t)bh * S_ + q0 + ln] = nm;
        srow[(size_t)bh * S_ + q0 + ln] = ns;
    }
}

// ---------------------------------------------------------------------------
// Attention pass 2: probabilities -> gate scores (atomics) + x^T = V^T @ W.
// Two consecutive 16x16 C-layout prob tiles == B-fragment for a K=32 WMMA.
__global__ __launch_bounds__(32, 1)
void attn_pass2(const __bf16* __restrict__ qb, const __bf16* __restrict__ kb,
                const __bf16* __restrict__ vt, const float* __restrict__ mrow,
                const float* __restrict__ srow, const int* __restrict__ lens,
                __bf16* __restrict__ xbf, float* __restrict__ scores) {
    int wid = blockIdx.x;                     // B*H*(S/16)
    int qt  = wid & (S_ / 16 - 1);
    int bh  = wid / (S_ / 16);
    int b   = bh / H_, h = bh % H_;
    int lane = threadIdx.x, hf = lane >> 4, ln = lane & 15;
    int q0 = qt * 16, len = lens[b];
    int qidx = q0 + ln;
    const __bf16* qrow = qb + ((size_t)bh * S_ + qidx) * HD_;
    bf16x16 bq0 = load_frag(qrow, 0,  hf);
    bf16x16 bq1 = load_frag(qrow, 32, hf);
    float mq = mrow[(size_t)bh * S_ + qidx];
    float inv_s = 1.0f / srow[(size_t)bh * S_ + qidx];
    float qgate = (qidx < len) ? (1.0f / (float)H_) : 0.0f;
    const __bf16* kbase = kb + (size_t)bh * S_ * HD_;
    f32x8 xacc[4] = {};
    for (int kc = 0; kc < S_; kc += 32) {
        f32x8 wsub[2];
#pragma unroll
        for (int st = 0; st < 2; ++st) {
            int kcc = kc + st * 16;
            const __bf16* krow = kbase + (size_t)(kcc + ln) * HD_;
            f32x8 acc = {};
            acc = wmma_bf16(load_frag(krow, 0,  hf), bq0, acc);
            acc = wmma_bf16(load_frag(krow, 32, hf), bq1, acc);
#pragma unroll
            for (int i = 0; i < 8; ++i) {
                int key = kcc + i + 8 * hf;
                float v = (key < len) ? __expf(acc[i] * 0.125f - mq) * inv_s : 0.0f;
                acc[i] = v;
            }
            wsub[st] = acc;
#pragma unroll
            for (int i = 0; i < 8; ++i) {   // gate scores: sum over 16 query cols
                float c = acc[i] * qgate;
                c += __shfl_xor(c, 1, 32);
                c += __shfl_xor(c, 2, 32);
                c += __shfl_xor(c, 4, 32);
                c += __shfl_xor(c, 8, 32);
                if (ln == 0)
                    atomicAdd(&scores[(size_t)b * S_ + kcc + i + 8 * hf], c);
            }
        }
        bf16x16 wb;
#pragma unroll
        for (int i = 0; i < 8; ++i) {
            wb[i]     = (__bf16)wsub[0][i];
            wb[i + 8] = (__bf16)wsub[1][i];
        }
#pragma unroll
        for (int t = 0; t < 4; ++t) {
            const __bf16* vrow = vt + ((size_t)bh * HD_ + t * 16 + ln) * S_;
            xacc[t] = wmma_bf16(load_frag(vrow, kc, hf), wb, xacc[t]);
        }
    }
#pragma unroll
    for (int t = 0; t < 4; ++t)
#pragma unroll
        for (int i = 0; i < 8; ++i) {
            int hd = t * 16 + i + 8 * hf;
            xbf[((size_t)b * S_ + qidx) * D_ + h * HD_ + hd] = (__bf16)xacc[t][i];
        }
}

// ---------------------------------------------------------------------------
// Per-sample top-k gate: bitonic sort in LDS (score desc, idx asc tiebreak),
// take STRIDE, invalidate i>=k_i, re-sort indices ascending.
__global__ void gate_topk(const float* __restrict__ scores, const unsigned char* __restrict__ pad,
                          int* __restrict__ gidx, float* __restrict__ validf,
                          float* __restrict__ ypad_out) {
    __shared__ float sc[S_];
    __shared__ int   si[S_];
    __shared__ int   red[1024];
    int b = blockIdx.x, t = threadIdx.x;
    red[t] = (pad[b * S_ + t] ? 1 : 0) + (pad[b * S_ + 1024 + t] ? 1 : 0);
    __syncthreads();
    for (int s = 512; s > 0; s >>= 1) { if (t < s) red[t] += red[t + s]; __syncthreads(); }
    int len = S_ - red[0];
    int k_i = max(1, (int)((float)len * 0.25f));
    for (int i = t; i < S_; i += 1024) {
        sc[i] = (i < len) ? scores[(size_t)b * S_ + i] : -1e30f;
        si[i] = i;
    }
    __syncthreads();
    for (int k = 2; k <= S_; k <<= 1)
        for (int j = k >> 1; j > 0; j >>= 1) {
            for (int idx = t; idx < S_; idx += 1024) {
                int p = idx ^ j;
                if (p > idx) {
                    bool up = ((idx & k) == 0);
                    float sa = sc[idx], sb = sc[p];
                    int   ia = si[idx], ib = si[p];
                    bool aBetter = (sa > sb) || (sa == sb && ia < ib);
                    if (up ? !aBetter : aBetter) {
                        sc[idx] = sb; si[idx] = ib; sc[p] = sa; si[p] = ia;
                    }
                }
            }
            __syncthreads();
        }
    if (t < STR_) si[t] = (t < k_i) ? si[t] : S_;
    __syncthreads();
    for (int k = 2; k <= STR_; k <<= 1)
        for (int j = k >> 1; j > 0; j >>= 1) {
            if (t < STR_) {
                int p = t ^ j;
                if (p > t && p < STR_) {
                    bool up = ((t & k) == 0);
                    int ia = si[t], ib = si[p];
                    if (up ? (ia > ib) : (ia < ib)) { si[t] = ib; si[p] = ia; }
                }
            }
            __syncthreads();
        }
    if (t < STR_) {
        int sv = si[t];
        gidx[b * STR_ + t] = sv > (S_ - 1) ? (S_ - 1) : sv;
        float v = (t < k_i) ? 1.0f : 0.0f;
        validf[b * STR_ + t]  = v;
        ypad_out[b * STR_ + t] = 1.0f - v;
    }
}

// ---------------------------------------------------------------------------
__global__ void gather_ln1(const float* __restrict__ src, const float* __restrict__ xproj,
                           const int* __restrict__ gidx, const float* __restrict__ validf,
                           const float* __restrict__ g, const float* __restrict__ bb,
                           float* __restrict__ y1, __bf16* __restrict__ y1b) {
    int row = blockIdx.x, t = threadIdx.x;   // row in [0, B*STR_)
    int b = row / STR_;
    int gi = gidx[row];
    float val = validf[row];
    const float* ss = src   + ((size_t)b * S_ + gi) * D_;
    const float* xs = xproj + ((size_t)b * S_ + gi) * D_;
    float v0 = (ss[t]       + xs[t])       * val;
    float v1 = (ss[t + 256] + xs[t + 256]) * val;
    __shared__ float r1[256], r2[256];
    r1[t] = v0 + v1; r2[t] = v0 * v0 + v1 * v1;
    __syncthreads();
    for (int s = 128; s > 0; s >>= 1) {
        if (t < s) { r1[t] += r1[t + s]; r2[t] += r2[t + s]; }
        __syncthreads();
    }
    float mean = r1[0] * (1.0f / D_);
    float var  = r2[0] * (1.0f / D_) - mean * mean;
    float inv  = rsqrtf(var + 1e-5f);
    float o0 = (v0 - mean) * inv * g[t]       + bb[t];
    float o1 = (v1 - mean) * inv * g[t + 256] + bb[t + 256];
    y1 [(size_t)row * D_ + t]       = o0;
    y1 [(size_t)row * D_ + t + 256] = o1;
    y1b[(size_t)row * D_ + t]       = (__bf16)o0;
    y1b[(size_t)row * D_ + t + 256] = (__bf16)o1;
}

__global__ void ln2_out(const float* __restrict__ y1, const float* __restrict__ h2,
                        const float* __restrict__ g, const float* __restrict__ bb,
                        float* __restrict__ out) {
    int row = blockIdx.x, t = threadIdx.x;
    float v0 = y1[(size_t)row * D_ + t]       + h2[(size_t)row * D_ + t];
    float v1 = y1[(size_t)row * D_ + t + 256] + h2[(size_t)row * D_ + t + 256];
    __shared__ float r1[256], r2[256];
    r1[t] = v0 + v1; r2[t] = v0 * v0 + v1 * v1;
    __syncthreads();
    for (int s = 128; s > 0; s >>= 1) {
        if (t < s) { r1[t] += r1[t + s]; r2[t] += r2[t + s]; }
        __syncthreads();
    }
    float mean = r1[0] * (1.0f / D_);
    float var  = r2[0] * (1.0f / D_) - mean * mean;
    float inv  = rsqrtf(var + 1e-5f);
    out[(size_t)row * D_ + t]       = (v0 - mean) * inv * g[t]       + bb[t];
    out[(size_t)row * D_ + t + 256] = (v1 - mean) * inv * g[t + 256] + bb[t + 256];
}

// ---------------------------------------------------------------------------
extern "C" void kernel_launch(void* const* d_in, const int* in_sizes, int n_in,
                              void* d_out, int out_size, void* d_ws, size_t ws_size,
                              hipStream_t stream) {
    const float* src   = (const float*)d_in[0];
    const unsigned char* spad = (const unsigned char*)d_in[1];
    const float* w_in  = (const float*)d_in[2];
    const float* b_in  = (const float*)d_in[3];
    const float* w_out = (const float*)d_in[4];
    const float* b_out = (const float*)d_in[5];
    const float* ln1_g = (const float*)d_in[6];
    const float* ln1_b = (const float*)d_in[7];
    const float* ln2_g = (const float*)d_in[8];
    const float* ln2_b = (const float*)d_in[9];
    const float* w1    = (const float*)d_in[10];
    const float* b1    = (const float*)d_in[11];
    const float* w2    = (const float*)d_in[12];
    const float* b2    = (const float*)d_in[13];
    float* out = (float*)d_out;

    uintptr_t p = (uintptr_t)d_ws;
    auto carve = [&](size_t bytes) -> void* {
        p = (p + 255) & ~(uintptr_t)255;
        void* r = (void*)p; p += bytes; return r;
    };
    const size_t NTOK = (size_t)B_ * S_;          // 8192
    __bf16* src_bf  = (__bf16*)carve(NTOK * D_ * 2);
    __bf16* w_in_bf = (__bf16*)carve((size_t)3 * D_ * D_ * 2);
    __bf16* w_out_bf= (__bf16*)carve((size_t)D_ * D_ * 2);
    __bf16* w1_bf   = (__bf16*)carve((size_t)FC_ * D_ * 2);
    __bf16* w2_bf   = (__bf16*)carve((size_t)D_ * FC_ * 2);
    float*  qkv     = (float*) carve(NTOK * 3 * D_ * 4);
    __bf16* qb      = (__bf16*)carve(NTOK * D_ * 2);
    __bf16* kb      = (__bf16*)carve(NTOK * D_ * 2);
    __bf16* vt      = (__bf16*)carve(NTOK * D_ * 2);
    float*  mrow    = (float*) carve((size_t)B_ * H_ * S_ * 4);
    float*  srow    = (float*) carve((size_t)B_ * H_ * S_ * 4);
    __bf16* xbf     = (__bf16*)carve(NTOK * D_ * 2);
    float*  scoresw = (float*) carve((size_t)B_ * S_ * 4);
    float*  xproj   = (float*) carve(NTOK * D_ * 4);
    int*    lens    = (int*)   carve(B_ * 4);
    int*    gidx    = (int*)   carve((size_t)B_ * STR_ * 4);
    float*  validf  = (float*) carve((size_t)B_ * STR_ * 4);
    float*  y1      = (float*) carve((size_t)B_ * STR_ * D_ * 4);
    __bf16* y1b     = (__bf16*)carve((size_t)B_ * STR_ * D_ * 2);
    float*  h1      = (float*) carve((size_t)B_ * STR_ * FC_ * 4);
    __bf16* h1b     = (__bf16*)carve((size_t)B_ * STR_ * FC_ * 2);
    float*  h2      = (float*) carve((size_t)B_ * STR_ * D_ * 4);

    auto cv = [&](const float* a, __bf16* o, int n) {
        cvt_bf16<<<(n + 255) / 256, 256, 0, stream>>>(a, o, n);
    };
    cv(src,   src_bf,  (int)(NTOK * D_));
    cv(w_in,  w_in_bf, 3 * D_ * D_);
    cv(w_out, w_out_bf, D_ * D_);
    cv(w1,    w1_bf,   FC_ * D_);
    cv(w2,    w2_bf,   D_ * FC_);

    // QKV projection: [8192,1536] = src_bf @ w_in^T + b_in
    gemm_bf16<<<dim3(3 * D_ / 64, NTOK / 256), 128, 0, stream>>>(
        src_bf, w_in_bf, b_in, qkv, (__bf16*)nullptr, (int)NTOK, 3 * D_, D_, 0);

    pack_qkv<<<(int)((NTOK * D_ + 255) / 256), 256, 0, stream>>>(qkv, qb, kb, vt);
    compute_lens<<<B_, 256, 0, stream>>>(spad, lens);
    zero_f32<<<(B_ * S_ + 255) / 256, 256, 0, stream>>>(scoresw, B_ * S_);

    int nqt = B_ * H_ * (S_ / 16);
    attn_stats<<<nqt, 32, 0, stream>>>(qb, kb, lens, mrow, srow);
    attn_pass2<<<nqt, 32, 0, stream>>>(qb, kb, vt, mrow, srow, lens, xbf, scoresw);

    // out projection: [8192,512] = xbf @ w_out^T + b_out
    gemm_bf16<<<dim3(D_ / 64, NTOK / 256), 128, 0, stream>>>(
        xbf, w_out_bf, b_out, xproj, (__bf16*)nullptr, (int)NTOK, D_, D_, 0);

    float* ypad_out = out + (size_t)B_ * STR_ * D_;
    gate_topk<<<B_, 1024, 0, stream>>>(scoresw, spad, gidx, validf, ypad_out);

    int nrow = B_ * STR_;  // 2048
    gather_ln1<<<nrow, 256, 0, stream>>>(src, xproj, gidx, validf, ln1_g, ln1_b, y1, y1b);

    // FFN1 (+SiLU): [2048,2048] = y1b @ w1^T + b1
    gemm_bf16<<<dim3(FC_ / 64, nrow / 256), 128, 0, stream>>>(
        y1b, w1_bf, b1, h1, h1b, nrow, FC_, D_, 1);
    // FFN2: [2048,512] = h1b @ w2^T + b2
    gemm_bf16<<<dim3(D_ / 64, nrow / 256), 128, 0, stream>>>(
        h1b, w2_bf, b2, h2, (__bf16*)nullptr, nrow, D_, FC_, 0);

    ln2_out<<<nrow, 256, 0, stream>>>(y1, h2, ln2_g, ln2_b, out);
}
